// InstantNGPMLP_63848983822902
// MI455X (gfx1250) — matrix operand order repaired and checked by reading, not compiled, not run on hardware
//
#include <hip/hip_runtime.h>

typedef _Float16 v16h  __attribute__((ext_vector_type(16)));
typedef float    v8f   __attribute__((ext_vector_type(8)));
typedef float    v4f   __attribute__((ext_vector_type(4)));
typedef _Float16 h2    __attribute__((ext_vector_type(2)));
typedef __fp16   fp16x2 __attribute__((ext_vector_type(2)));

struct Pack8 { unsigned u[8]; };

static __device__ __forceinline__ v16h toV16(const unsigned (&u)[8]) {
  Pack8 p;
#pragma unroll
  for (int i = 0; i < 8; ++i) p.u[i] = u[i];
  return __builtin_bit_cast(v16h, p);
}

// single-instruction packed f32->f16 convert (v_cvt_pk_rtz_f16_f32)
static __device__ __forceinline__ unsigned pkf(float a, float b) {
  fp16x2 v = __builtin_amdgcn_cvt_pkrtz(a, b);
  return __builtin_bit_cast(unsigned, v);
}

// convert two f32 accumulator values to packed f16, add packed-f16 bias, ReLU
static __device__ __forceinline__ unsigned cvt_bias_relu(float a, float b, unsigned biasPk) {
  fp16x2 t = __builtin_amdgcn_cvt_pkrtz(a, b);
  h2 v  = __builtin_bit_cast(h2, t);
  h2 bb = __builtin_bit_cast(h2, biasPk);
  v = v + bb;                                        // v_pk_add_f16
  v[0] = (v[0] > (_Float16)0) ? v[0] : (_Float16)0;
  v[1] = (v[1] > (_Float16)0) ? v[1] : (_Float16)0;  // v_pk_max_num_f16
  return __builtin_bit_cast(unsigned, v);
}

// A-matrix (16x32 f16) element i -> K index, contiguous order: K=16h+i (matches B fragments)
static __device__ __forceinline__ int kIdx(int i, int h) {
  return 16 * h + i;
}

#define WMMA_F16(a, b, c) \
  __builtin_amdgcn_wmma_f32_16x16x32_f16(false, (a), false, (b), (short)0, (c), false, false)

// D tiles (packed f16 pairs P[t][j]: feats 16t+8h+2j,+1 of sample l) -> B fragments for next
// layer (2 K-chunks of 32). Only a lane-xor-16 swap is needed:
//   B_c.Vp     (p<4) = h==0 ? P[2c][p]          : shfl16(P[2c+1][p])
//   B_c.V(p+4) (p<4) = h==0 ? shfl16(P[2c][p])  : P[2c+1][p]
static __device__ __forceinline__ void buildB(const unsigned (&P)[4][4], int h, v16h (&B)[2]) {
#pragma unroll
  for (int c = 0; c < 2; ++c) {
    unsigned u[8];
#pragma unroll
    for (int p = 0; p < 4; ++p) {
      unsigned d0 = P[2 * c][p];
      unsigned d1 = P[2 * c + 1][p];
      unsigned s1 = (unsigned)__shfl_xor((int)d1, 16, 32);
      unsigned s0 = (unsigned)__shfl_xor((int)d0, 16, 32);
      u[p]     = h ? s1 : d0;
      u[p + 4] = h ? d1 : s0;
    }
    B[c] = toV16(u);
  }
}

__global__ void __launch_bounds__(256, 1)
mlp_wmma_kernel(const float* __restrict__ x,
                const float* __restrict__ Win, const float* __restrict__ bin,
                const float* __restrict__ W1,  const float* __restrict__ b1,
                const float* __restrict__ W2,  const float* __restrict__ b2,
                const float* __restrict__ Wout, const float* __restrict__ bout,
                float* __restrict__ out, int nTiles) {
  const int lane = threadIdx.x & 31;
  const int l    = lane & 15;   // sample-in-tile (B/D column)
  const int h    = lane >> 4;   // half-wave select
  const int waveId = (int)((blockIdx.x * blockDim.x + threadIdx.x) >> 5);
  const int nWaves = (int)((gridDim.x * blockDim.x) >> 5);

  // ---- one-time per-wave weight gather into register-resident A fragments (f16) ----
  // Transposed formulation: A_t[m][k] = W[k][16t+m]  (W stored fan_in x fan_out row-major)
  v16h A1[4];
#pragma unroll
  for (int t = 0; t < 4; ++t) {
    const int f = 16 * t + l;
#pragma unroll
    for (int i = 0; i < 16; ++i)
      A1[t][i] = (_Float16)Win[kIdx(i, h) * 64 + f];
  }
  v16h A2[4][2];
#pragma unroll
  for (int t = 0; t < 4; ++t) {
    const int f = 16 * t + l;
#pragma unroll
    for (int c = 0; c < 2; ++c)
#pragma unroll
      for (int i = 0; i < 16; ++i)
        A2[t][c][i] = (_Float16)W1[(32 * c + kIdx(i, h)) * 64 + f];
  }
  v16h A3[4][2];
#pragma unroll
  for (int t = 0; t < 4; ++t) {
    const int f = 16 * t + l;
#pragma unroll
    for (int c = 0; c < 2; ++c)
#pragma unroll
      for (int i = 0; i < 16; ++i)
        A3[t][c][i] = (_Float16)W2[(32 * c + kIdx(i, h)) * 64 + f];
  }
  v16h A4[2];
#pragma unroll
  for (int c = 0; c < 2; ++c)
#pragma unroll
    for (int i = 0; i < 16; ++i)
      A4[c][i] = (_Float16)Wout[(32 * c + kIdx(i, h)) * 16 + l];

  // ---- biases packed to match D layout: acc[2j] is out-feature 16t+8h+2j ----
  unsigned BP1[4][4], BP2[4][4], BP3[4][4];
#pragma unroll
  for (int t = 0; t < 4; ++t)
#pragma unroll
    for (int j = 0; j < 4; ++j) {
      const int base = 16 * t + 8 * h + 2 * j;
      BP1[t][j] = pkf(bin[base], bin[base + 1]);
      BP2[t][j] = pkf(b1[base],  b1[base + 1]);
      BP3[t][j] = pkf(b2[base],  b2[base + 1]);
    }
  float BO[8];
#pragma unroll
  for (int r = 0; r < 8; ++r) BO[r] = bout[r + 8 * h];

  if (nTiles <= 0) return;
  const int laneOff = l * 32 + 16 * h;

  // ---- software-pipelined main loop: prefetch next tile's x behind the WMMA chain ----
  int tile = waveId;
  v4f q0 = {}, q1 = {}, q2 = {}, q3 = {};
  if (tile < nTiles) {
    const v4f* xp = (const v4f*)(x + (size_t)tile * (16 * 32) + laneOff);
    q0 = __builtin_nontemporal_load(xp + 0);
    q1 = __builtin_nontemporal_load(xp + 1);
    q2 = __builtin_nontemporal_load(xp + 2);
    q3 = __builtin_nontemporal_load(xp + 3);
  }

  for (; tile < nTiles; tile += nWaves) {
    // B1 = X^T fragment: lane (l,h) holds features 16h..16h+15 of sample tile*16+l
    unsigned ub[8] = {pkf(q0[0], q0[1]), pkf(q0[2], q0[3]), pkf(q1[0], q1[1]), pkf(q1[2], q1[3]),
                      pkf(q2[0], q2[1]), pkf(q2[2], q2[3]), pkf(q3[0], q3[1]), pkf(q3[2], q3[3])};
    const v16h B1 = toV16(ub);

    // issue next tile's loads now; they complete behind the 4-layer compute chain
    {
      const int nt   = tile + nWaves;
      const int safe = (nt < nTiles) ? nt : tile;
      const v4f* xp = (const v4f*)(x + (size_t)safe * (16 * 32) + laneOff);
      q0 = __builtin_nontemporal_load(xp + 0);
      q1 = __builtin_nontemporal_load(xp + 1);
      q2 = __builtin_nontemporal_load(xp + 2);
      q3 = __builtin_nontemporal_load(xp + 3);
    }

    // Layer 1: 4 out-feature tiles, K=32
    unsigned P[4][4];
#pragma unroll
    for (int t = 0; t < 4; ++t) {
      v8f acc = {};
      acc = WMMA_F16(A1[t], B1, acc);
#pragma unroll
      for (int j = 0; j < 4; ++j)
        P[t][j] = cvt_bias_relu(acc[2 * j], acc[2 * j + 1], BP1[t][j]);
    }

    // Layer 2: K=64 (2 chunks)
    v16h Bc[2];
    buildB(P, h, Bc);
#pragma unroll
    for (int t = 0; t < 4; ++t) {
      v8f acc = {};
      acc = WMMA_F16(A2[t][0], Bc[0], acc);
      acc = WMMA_F16(A2[t][1], Bc[1], acc);
#pragma unroll
      for (int j = 0; j < 4; ++j)
        P[t][j] = cvt_bias_relu(acc[2 * j], acc[2 * j + 1], BP2[t][j]);
    }

    // Layer 3
    buildB(P, h, Bc);
#pragma unroll
    for (int t = 0; t < 4; ++t) {
      v8f acc = {};
      acc = WMMA_F16(A3[t][0], Bc[0], acc);
      acc = WMMA_F16(A3[t][1], Bc[1], acc);
#pragma unroll
      for (int j = 0; j < 4; ++j)
        P[t][j] = cvt_bias_relu(acc[2 * j], acc[2 * j + 1], BP3[t][j]);
    }

    // Layer 4: 16 out features = 1 tile, K=64, f32 bias, no ReLU
    buildB(P, h, Bc);
    v8f acc = {};
    acc = WMMA_F16(A4[0], Bc[0], acc);
    acc = WMMA_F16(A4[1], Bc[1], acc);

    // D layout: acc[r] = out-feature r+8h of sample tile*16+l -> contiguous 32B per lane
    float* op = out + ((size_t)tile * 16 + l) * 16 + 8 * h;
    v4f o0 = {acc[0] + BO[0], acc[1] + BO[1], acc[2] + BO[2], acc[3] + BO[3]};
    v4f o1 = {acc[4] + BO[4], acc[5] + BO[5], acc[6] + BO[6], acc[7] + BO[7]};
    __builtin_nontemporal_store(o0, (v4f*)op + 0);
    __builtin_nontemporal_store(o1, (v4f*)op + 1);
  }
}

extern "C" void kernel_launch(void* const* d_in, const int* in_sizes, int n_in,
                              void* d_out, int out_size, void* d_ws, size_t ws_size,
                              hipStream_t stream) {
  const float* x    = (const float*)d_in[0];
  const float* Win  = (const float*)d_in[1];
  const float* bin  = (const float*)d_in[2];
  const float* W1   = (const float*)d_in[3];
  const float* b1   = (const float*)d_in[4];
  const float* W2   = (const float*)d_in[5];
  const float* b2   = (const float*)d_in[6];
  const float* Wout = (const float*)d_in[7];
  const float* bout = (const float*)d_in[8];
  float* out = (float*)d_out;

  const int N = in_sizes[0] / 32;   // rows
  const int nTiles = N / 16;        // N = 1048576 -> 65536 tiles

  const int threads = 256;          // 8 wave32s per block
  const int blocks  = 1024;         // 8192 waves -> 8 tiles per wave (amortizes weight gather)
  mlp_wmma_kernel<<<blocks, threads, 0, stream>>>(x, Win, bin, W1, b1, W2, b2, Wout, bout,
                                                  out, nTiles);
}
